// SparseAutoencoder_86698209837598
// MI455X (gfx1250) — compile-verified, matrix operand order checked by reading, not run on hardware
//
#include <hip/hip_runtime.h>

typedef __attribute__((ext_vector_type(16))) __bf16 v16bf;
typedef __attribute__((ext_vector_type(8)))  __bf16 v8bf;
typedef __attribute__((ext_vector_type(8)))  float  v8f;

#define KTOP 64

// ---- gfx1250 async copy: global -> LDS, 16B per lane, tracked by ASYNCcnt ----
__device__ __forceinline__ void async_copy_b128(unsigned lds_byte_addr,
                                                unsigned global_byte_off,
                                                const void* base) {
  asm volatile("global_load_async_to_lds_b128 %0, %1, %2"
               :
               : "v"(lds_byte_addr), "v"(global_byte_off), "s"(base)
               : "memory");
}
__device__ __forceinline__ void wait_async0() {
  asm volatile("s_wait_asynccnt 0x0" ::: "memory");
}

// ---------------- f32 -> bf16 pack (8 elems / thread) ----------------
__global__ __launch_bounds__(256) void cvt_f32_bf16(const float* __restrict__ in,
                                                    __bf16* __restrict__ out, long n8) {
  long i = (long)blockIdx.x * blockDim.x + threadIdx.x;
  if (i >= n8) return;
  const float4* p = (const float4*)in + i * 2;
  float4 a = p[0], b = p[1];
  v8bf o;
  o[0] = (__bf16)a.x; o[1] = (__bf16)a.y; o[2] = (__bf16)a.z; o[3] = (__bf16)a.w;
  o[4] = (__bf16)b.x; o[5] = (__bf16)b.y; o[6] = (__bf16)b.z; o[7] = (__bf16)b.w;
  ((v8bf*)out)[i] = o;
}

// ---------------- dec_w [D,H] f32 -> [H,D] bf16 transpose ----------------
__global__ __launch_bounds__(256) void transpose_to_bf16(const float* __restrict__ in,
                                                         __bf16* __restrict__ out,
                                                         int Dd, int Hd) {
  __shared__ float tile[32][33];
  const int hx = blockIdx.x * 32, dx = blockIdx.y * 32;
  const int tx = threadIdx.x, ty = threadIdx.y;  // (32,8)
  #pragma unroll
  for (int r = 0; r < 32; r += 8)
    tile[ty + r][tx] = in[(long)(dx + ty + r) * Hd + hx + tx];
  __syncthreads();
  #pragma unroll
  for (int r = 0; r < 32; r += 8)
    out[(long)(hx + ty + r) * Dd + dx + tx] = (__bf16)tile[tx][ty + r];
}

// ---------------- encoder GEMM: pre = relu(x @ Wᵀ + b), WMMA bf16 ----------------
// Block: 128(M) x 256(N) tile, 8 waves in 2x4; each wave: 4x4 16x16 tiles (64x64).
// Tiles staged global->LDS with GLOBAL_LOAD_ASYNC_TO_LDS_B128, double buffered.
__global__ __launch_bounds__(256) void enc_gemm_wmma(
    const __bf16* __restrict__ X,    // [B,D] row-major
    const __bf16* __restrict__ W,    // [H,D] row-major
    const float*  __restrict__ bias, // [H]
    float* __restrict__ Hout,        // [B,H]
    int Dd, int Hd) {
  __shared__ alignas(16) __bf16 xs[2][128 * 32];   //  8 KB per stage
  __shared__ alignas(16) __bf16 wsm[2][256 * 32];  // 16 KB per stage

  const int tid  = threadIdx.x;
  const int lane = tid & 31, wave = tid >> 5;
  const int wm = wave & 1;   // 0..1 -> M offset wm*64
  const int wn = wave >> 1;  // 0..3 -> N offset wn*64
  const int hi = lane >> 4;  // K-half select per ISA fragment layout
  const int lr = lane & 15;

  const long mtile = (long)blockIdx.y * 128;
  const long ntile = (long)blockIdx.x * 256;

  // staging geometry: 16B segment per lane; seg s -> row s>>2, col8 (s&3)*8
  const int r0 = tid >> 2;        // 0..63
  const int c0 = (tid & 3) * 8;   // 0,8,16,24

  // LDS byte addresses (AS(3) offset == low 32 bits of generic pointer)
  const unsigned xsBase = (unsigned)(size_t)(&xs[0][0]);
  const unsigned wsBase = (unsigned)(size_t)(&wsm[0][0]);

  // global byte offsets (k=0), advance by kt*64 bytes each stage
  const unsigned xg0 = (unsigned)(((mtile + r0) * Dd + c0) * 2);
  const unsigned xg1 = (unsigned)(((mtile + r0 + 64) * Dd + c0) * 2);
  const unsigned wg0 = (unsigned)(((ntile + r0) * Dd + c0) * 2);
  const unsigned wg1 = (unsigned)(((ntile + r0 + 64) * Dd + c0) * 2);
  const unsigned wg2 = (unsigned)(((ntile + r0 + 128) * Dd + c0) * 2);
  const unsigned wg3 = (unsigned)(((ntile + r0 + 192) * Dd + c0) * 2);

  auto issue_stage = [&](int kt, int nb) {
    const unsigned kb = (unsigned)kt * 64u;  // 32 k-elems * 2 bytes
    const unsigned xl = xsBase + (unsigned)nb * 8192u + (unsigned)tid * 16u;
    async_copy_b128(xl,         xg0 + kb, X);
    async_copy_b128(xl + 4096u, xg1 + kb, X);
    const unsigned wl = wsBase + (unsigned)nb * 16384u + (unsigned)tid * 16u;
    async_copy_b128(wl,          wg0 + kb, W);
    async_copy_b128(wl + 4096u,  wg1 + kb, W);
    async_copy_b128(wl + 8192u,  wg2 + kb, W);
    async_copy_b128(wl + 12288u, wg3 + kb, W);
  };

  v8f acc[4][4];
  #pragma unroll
  for (int i = 0; i < 4; ++i)
    #pragma unroll
    for (int j = 0; j < 4; ++j) {
      v8f z = {};
      acc[i][j] = z;
    }

  const int NK = Dd >> 5;  // K-steps of 32

  issue_stage(0, 0);
  wait_async0();
  __syncthreads();

  for (int kt = 0; kt < NK; ++kt) {
    const int buf = kt & 1;
    if (kt + 1 < NK) issue_stage(kt + 1, buf ^ 1);  // target buffer freed by last iter's barrier

    // Materialize ALL fragments first (8 independent register octets) so the
    // 16 ds_load_b128 issue up-front and s_wait_dscnt staggers, instead of a
    // full drain before each WMMA group.
    v16bf bfrag[4], afrag[4];
    #pragma unroll
    for (int j = 0; j < 4; ++j) {
      const v8bf* p = (const v8bf*)(wsm[buf] + (wn * 64 + j * 16 + lr) * 32);
      v8bf lo = p[hi], hh = p[2 + hi];
      bfrag[j] = __builtin_shufflevector(lo, hh, 0, 1, 2, 3, 4, 5, 6, 7, 8, 9, 10, 11, 12, 13, 14, 15);
    }
    #pragma unroll
    for (int i = 0; i < 4; ++i) {
      const v8bf* p = (const v8bf*)(xs[buf] + (wm * 64 + i * 16 + lr) * 32);
      v8bf lo = p[hi], hh = p[2 + hi];
      afrag[i] = __builtin_shufflevector(lo, hh, 0, 1, 2, 3, 4, 5, 6, 7, 8, 9, 10, 11, 12, 13, 14, 15);
    }

    #pragma unroll
    for (int i = 0; i < 4; ++i)
      #pragma unroll
      for (int j = 0; j < 4; ++j)
        acc[i][j] = __builtin_amdgcn_wmma_f32_16x16x32_bf16(
            false, afrag[i], false, bfrag[j], (short)0, acc[i][j], false, false);

    __syncthreads();  // all waves done reading buf
    if (kt + 1 < NK) {
      wait_async0();     // my async writes into buf^1 landed in LDS
      __syncthreads();   // everyone's landed
    }
  }

  // epilogue: C/D layout — VGPR v holds M = v + 8*hi, N = lr
  #pragma unroll
  for (int j = 0; j < 4; ++j) {
    const long n = ntile + wn * 64 + j * 16 + lr;
    const float bv = bias[n];
    #pragma unroll
    for (int i = 0; i < 4; ++i) {
      const long row0 = mtile + wm * 64 + i * 16 + hi * 8;
      #pragma unroll
      for (int v = 0; v < 8; ++v) {
        float val = acc[i][j][v] + bv;
        val = val > 0.f ? val : 0.f;
        Hout[(row0 + v) * Hd + n] = val;
      }
    }
  }
}

// ---------------- per-row top-64: radix-select on float bits ----------------
// Post-ReLU values are >= 0, so uint order == float order. Deterministic ties
// (lowest index wins) via chunked prefix sums.
__global__ __launch_bounds__(256) void topk_select(float* __restrict__ Hrow,
                                                   float* __restrict__ topv,
                                                   int* __restrict__ topi, int Hd) {
  const int b = blockIdx.x, tid = threadIdx.x;
  float* row = Hrow + (long)b * Hd;

  __shared__ unsigned hist[256];
  __shared__ unsigned s_prefix;
  __shared__ int s_want;
  __shared__ int cgt[256], ceq[256];

  unsigned prefix = 0;
  int want = KTOP;
  for (int p = 3; p >= 0; --p) {
    hist[tid] = 0;
    __syncthreads();
    const unsigned mask = (p == 3) ? 0u : (0xFFFFFFFFu << ((p + 1) * 8));
    for (int i = tid; i < Hd; i += 256) {
      unsigned u = __float_as_uint(row[i]);
      if ((u & mask) == (prefix & mask))
        atomicAdd(&hist[(u >> (p * 8)) & 0xFFu], 1u);
    }
    __syncthreads();
    if (tid == 0) {
      int above = 0, bsel = 0;
      for (int bi = 255; bi >= 0; --bi) {
        int c = (int)hist[bi];
        if (above + c >= want) { bsel = bi; break; }
        above += c;
      }
      s_prefix = prefix | ((unsigned)bsel << (p * 8));
      s_want = want - above;
    }
    __syncthreads();
    prefix = s_prefix;
    want = s_want;
    __syncthreads();
  }
  const unsigned thr = prefix;   // bits of the 64th-largest value
  const int quota = want;        // #equals to keep
  const int ngt = KTOP - quota;  // #strictly greater

  const int CH = Hd / 256;
  const int beg = tid * CH, end = beg + CH;
  int lgt = 0, leq = 0;
  for (int i = beg; i < end; ++i) {
    unsigned u = __float_as_uint(row[i]);
    if (u > thr) lgt++; else if (u == thr) leq++;
  }
  cgt[tid] = lgt; ceq[tid] = leq;
  __syncthreads();
  int pgt = 0, peq = 0;
  for (int j = 0; j < tid; ++j) { pgt += cgt[j]; peq += ceq[j]; }
  for (int i = beg; i < end; ++i) {
    float v = row[i];
    unsigned u = __float_as_uint(v);
    if (u > thr) {
      topv[(long)b * KTOP + pgt] = v;
      topi[(long)b * KTOP + pgt] = i;
      pgt++;
    } else if (u == thr) {
      if (peq < quota) {
        int s = ngt + peq;
        topv[(long)b * KTOP + s] = v;
        topi[(long)b * KTOP + s] = i;
      } else {
        row[i] = 0.f;
      }
      peq++;
    } else {
      row[i] = 0.f;
    }
  }
}

// ---------------- sparse decoder: x_hat[b,:] = sum_j h[b,j]*WT[j,:] + dec_b ----------------
__global__ __launch_bounds__(256) void decode_sparse(const float* __restrict__ topv,
                                                     const int* __restrict__ topi,
                                                     const __bf16* __restrict__ WT,  // [H,D]
                                                     const float* __restrict__ db,   // [D]
                                                     float* __restrict__ Xhat,       // [B,D]
                                                     int Dd) {
  const int b = blockIdx.x, tid = threadIdx.x;
  __shared__ float sv[KTOP];
  __shared__ int si[KTOP];
  if (tid < KTOP) {
    sv[tid] = topv[(long)b * KTOP + tid];
    si[tid] = topi[(long)b * KTOP + tid];
  }
  __syncthreads();

  float acc[16];
  const float4* bp = (const float4*)(db + tid * 16);
  #pragma unroll
  for (int q = 0; q < 4; ++q) {
    float4 t = bp[q];
    acc[q * 4 + 0] = t.x; acc[q * 4 + 1] = t.y;
    acc[q * 4 + 2] = t.z; acc[q * 4 + 3] = t.w;
  }
  for (int nz = 0; nz < KTOP; ++nz) {
    const float v = sv[nz];
    const v8bf* wr = (const v8bf*)(WT + (long)si[nz] * Dd + tid * 16);
    v8bf w0 = wr[0], w1 = wr[1];
    #pragma unroll
    for (int e = 0; e < 8; ++e) {
      acc[e]     += v * (float)w0[e];
      acc[8 + e] += v * (float)w1[e];
    }
  }
  float4* op = (float4*)(Xhat + (long)b * Dd + tid * 16);
  #pragma unroll
  for (int q = 0; q < 4; ++q) {
    float4 t;
    t.x = acc[q * 4 + 0]; t.y = acc[q * 4 + 1];
    t.z = acc[q * 4 + 2]; t.w = acc[q * 4 + 3];
    op[q] = t;
  }
}

extern "C" void kernel_launch(void* const* d_in, const int* in_sizes, int n_in,
                              void* d_out, int out_size, void* d_ws, size_t ws_size,
                              hipStream_t stream) {
  const float* x     = (const float*)d_in[0];  // [B,D]
  const float* enc_w = (const float*)d_in[1];  // [H,D]
  const float* enc_b = (const float*)d_in[2];  // [H]
  const float* dec_w = (const float*)d_in[3];  // [D,H]
  const float* dec_b = (const float*)d_in[4];  // [D]

  const int Hd = in_sizes[2];
  const int Dd = in_sizes[4];
  const int Bb = (int)((long)in_sizes[0] / Dd);

  const long nBD = (long)Bb * Dd;
  const long nHD = (long)Hd * Dd;

  char* ws = (char*)d_ws;
  __bf16* xb   = (__bf16*)ws;                              // B*D bf16
  __bf16* wb   = (__bf16*)(ws + nBD * 2);                  // H*D bf16
  __bf16* wtb  = (__bf16*)(ws + nBD * 2 + nHD * 2);        // H*D bf16 (dec_wT)
  float*  topv = (float*)(ws + nBD * 2 + nHD * 4);         // B*64 f32
  int*    topi = (int*)(ws + nBD * 2 + nHD * 4 + (long)Bb * KTOP * 4);

  float* hout = (float*)d_out;            // [B,H]
  float* xhat = hout + (long)Bb * Hd;     // [B,D]

  // 1) pack activations + encoder weights to bf16
  cvt_f32_bf16<<<(int)((nBD / 8 + 255) / 256), 256, 0, stream>>>(x, xb, nBD / 8);
  cvt_f32_bf16<<<(int)((nHD / 8 + 255) / 256), 256, 0, stream>>>(enc_w, wb, nHD / 8);

  // 2) transpose decoder weights to [H,D] bf16 (fits in 192MB L2)
  dim3 tg(Hd / 32, Dd / 32);
  transpose_to_bf16<<<tg, dim3(32, 8), 0, stream>>>(dec_w, wtb, Dd, Hd);

  // 3) encoder GEMM + bias + ReLU (WMMA bf16, f32 accumulate, async LDS staging)
  dim3 gg(Hd / 256, Bb / 128);
  enc_gemm_wmma<<<gg, 256, 0, stream>>>(xb, wb, enc_b, hout, Dd, Hd);

  // 4) top-64 per row: zero the rest, emit compact (idx,val) lists
  topk_select<<<Bb, 256, 0, stream>>>(hout, topv, topi, Hd);

  // 5) sparse decode + decoder bias
  decode_sparse<<<Bb, 256, 0, stream>>>(topv, topi, wtb, dec_b, xhat, Dd);
}
